// scGAAC_32435593020024
// MI455X (gfx1250) — compile-verified, matrix operand order checked by reading, not run on hardware
//
#include <hip/hip_runtime.h>
#include <hip/hip_bf16.h>
#include <math.h>

// ---------------- types / constants ----------------
typedef __attribute__((ext_vector_type(16))) __bf16 bf16x16;
typedef __attribute__((ext_vector_type(8)))  float  f32x8;

#define NN   4096
#define N_IN 3000
#define E1D  500
#define E2D  500
#define E3D  2000
#define NZD  10
#define NCD  10

enum { ACT_NONE = 0, ACT_RELU = 1, ACT_ELU = 2, ACT_SIG = 3 };

__device__ __forceinline__ unsigned short f2bf(float f) {
  union { float f; unsigned int u; } c; c.f = f;
  unsigned int u = c.u;
  u += 0x7fffu + ((u >> 16) & 1u);   // round-to-nearest-even
  return (unsigned short)(u >> 16);
}

__device__ __forceinline__ unsigned int pack2bf(float lo, float hi) {
  return (unsigned int)f2bf(lo) | ((unsigned int)f2bf(hi) << 16);
}

__device__ __forceinline__ float apply_act(float v, int act) {
  if (act == ACT_RELU) return v > 0.0f ? v : 0.0f;
  if (act == ACT_ELU)  return v > 0.0f ? v : (expf(v) - 1.0f);
  if (act == ACT_SIG)  return 1.0f / (1.0f + expf(-v));
  return v;
}

// ---------------- WMMA bf16 GEMM: C = act(A@B + bias) ----------------
// A: [M,K] f32 row-major (M must be a multiple of 128 — always 4096 here).
// B: [K,N] f32 row-major, or if transB: B(k,n) = Bp[n*ldb + k].
#define BM 128
#define BN 64
#define BKK 32
#define LK 40   // padded LDS row stride in ushorts (80 B: 16B-aligned, conflict-free)

__global__ __launch_bounds__(256) void gemm_bf16_wmma(
    const float* __restrict__ A, const float* __restrict__ B,
    const float* __restrict__ bias, float* __restrict__ C,
    int M, int N, int K, int ldb, int transB, int act)
{
  __shared__ unsigned short As[BM * LK];
  __shared__ unsigned short Bs[BN * LK];

  const int tid  = threadIdx.x;
  const int lane = tid & 31;
  const int wave = tid >> 5;       // 0..7
  const int wm   = wave >> 1;      // 0..3 : rows [wm*32, wm*32+32)
  const int wn   = wave & 1;       // 0..1 : cols [wn*32, wn*32+32)
  const int lh   = lane >> 4;      // lane half
  const int ml   = lane & 15;
  const int row0 = blockIdx.y * BM;
  const int col0 = blockIdx.x * BN;
  const size_t lda = (size_t)K;
  const size_t ldc = (size_t)N;
  const size_t ldbs = (size_t)ldb;

  // ---- hoisted per-thread staging geometry ----
  // A: thread handles rows rA+16i (i<8), k-pair kpA
  const int rA  = tid >> 4;
  const int kpA = (tid & 15) * 2;
  const float* abase = A + (size_t)(row0 + rA) * lda + kpA;
  const size_t astr = (size_t)16 * lda;
  const int aw = rA * LK + kpA;          // LDS write base (advance 16*LK per i)

  // B (!transB): n fixed = tid&63, k-pair = (tid>>6)*2 + 8i (i<4)
  const int nB0  = tid & 63;
  const int kpB0 = (tid >> 6) * 2;
  const int gnB0 = col0 + nB0;
  const float mB0 = (gnB0 < N) ? 1.0f : 0.0f;
  const float* bbase0 = B + (size_t)kpB0 * ldbs + (size_t)(gnB0 < N ? gnB0 : 0);
  const size_t bstr0 = (size_t)8 * ldbs;
  const int bw0 = nB0 * LK + kpB0;       // advance 8 per i

  // B (transB): k-pair fixed = (tid&15)*2, n = (tid>>4) + 16i (i<4)
  const int kpB1 = (tid & 15) * 2;
  const int nB1  = tid >> 4;
  const float* bbase1[4];
  float mB1[4];
#pragma unroll
  for (int i = 0; i < 4; ++i) {
    int gn = col0 + nB1 + i * 16;
    mB1[i] = (gn < N) ? 1.0f : 0.0f;
    bbase1[i] = B + (size_t)(gn < N ? gn : 0) * ldbs + kpB1;
  }
  const int bw1 = nB1 * LK + kpB1;       // advance 16*LK per i

  f32x8 acc00 = {}, acc01 = {}, acc10 = {}, acc11 = {};

  for (int k0 = 0; k0 < K; k0 += BKK) {
    const bool full = (k0 + BKK) <= K;   // uniform across block
    if (full) {
      // ---- HOT PATH: batched unguarded loads, then convert+store ----
      float av[16];
#pragma unroll
      for (int i = 0; i < 8; ++i) {
        const float* p = abase + (size_t)k0 + (size_t)i * astr;
        av[2 * i]     = p[0];
        av[2 * i + 1] = p[1];
      }
      float bv[8];
      if (!transB) {
        const float* q = bbase0 + (size_t)k0 * ldbs;
#pragma unroll
        for (int i = 0; i < 4; ++i) {
          const float* p = q + (size_t)i * bstr0;
          bv[2 * i]     = p[0];
          bv[2 * i + 1] = p[ldbs];
        }
      } else {
#pragma unroll
        for (int i = 0; i < 4; ++i) {
          const float* p = bbase1[i] + (size_t)k0;
          bv[2 * i]     = p[0];
          bv[2 * i + 1] = p[1];
        }
      }
#pragma unroll
      for (int i = 0; i < 8; ++i)
        *(unsigned int*)&As[aw + i * (16 * LK)] = pack2bf(av[2 * i], av[2 * i + 1]);
      if (!transB) {
#pragma unroll
        for (int i = 0; i < 4; ++i)
          *(unsigned int*)&Bs[bw0 + i * 8] = pack2bf(bv[2 * i] * mB0, bv[2 * i + 1] * mB0);
      } else {
#pragma unroll
        for (int i = 0; i < 4; ++i)
          *(unsigned int*)&Bs[bw1 + i * (16 * LK)] = pack2bf(bv[2 * i] * mB1[i], bv[2 * i + 1] * mB1[i]);
      }
    } else {
      // ---- K-tail (runs at most once per GEMM): guarded branchless ----
#pragma unroll
      for (int i = 0; i < 8; ++i) {
        int gk = k0 + kpA;
        const float* p = abase + (size_t)i * astr;
        bool ok0 = gk < K, ok1 = (gk + 1) < K;
        float v0 = p[ok0 ? (size_t)k0 : 0];       v0 = ok0 ? v0 : 0.0f;
        float v1 = p[ok1 ? (size_t)(k0 + 1) : 0]; v1 = ok1 ? v1 : 0.0f;
        *(unsigned int*)&As[aw + i * (16 * LK)] = pack2bf(v0, v1);
      }
      if (!transB) {
#pragma unroll
        for (int i = 0; i < 4; ++i) {
          int gk = k0 + kpB0 + i * 8;
          bool ok0 = (gk < K), ok1 = ((gk + 1) < K);
          const float* p = bbase0 + (size_t)i * bstr0;
          float v0 = p[ok0 ? (size_t)k0 * ldbs : 0];        v0 = ok0 ? v0 * mB0 : 0.0f;
          float v1 = p[ok1 ? ((size_t)k0 * ldbs + ldbs) : 0]; v1 = ok1 ? v1 * mB0 : 0.0f;
          *(unsigned int*)&Bs[bw0 + i * 8] = pack2bf(v0, v1);
        }
      } else {
#pragma unroll
        for (int i = 0; i < 4; ++i) {
          int gk = k0 + kpB1;
          bool ok0 = gk < K, ok1 = (gk + 1) < K;
          const float* p = bbase1[i];
          float v0 = p[ok0 ? (size_t)k0 : 0];       v0 = ok0 ? v0 * mB1[i] : 0.0f;
          float v1 = p[ok1 ? (size_t)(k0 + 1) : 0]; v1 = ok1 ? v1 * mB1[i] : 0.0f;
          *(unsigned int*)&Bs[bw1 + i * (16 * LK)] = pack2bf(v0, v1);
        }
      }
    }
    // prefetch next A k-tile
    if (k0 + BKK < K) __builtin_prefetch(abase + (size_t)(k0 + BKK), 0, 1);
    __syncthreads();

    // ---- fragments: element j -> K = (j<8?0:16) + lh*8 + (j&7); lane -> M/N = lane&15 ----
    union Frag { bf16x16 v; unsigned short u[16]; };
    Frag fa0, fa1, fb0, fb1;
    {
      const unsigned short* a0 = &As[(wm * 32 + ml) * LK];
      const unsigned short* a1 = &As[(wm * 32 + 16 + ml) * LK];
      const unsigned short* b0 = &Bs[(wn * 32 + ml) * LK];
      const unsigned short* b1 = &Bs[(wn * 32 + 16 + ml) * LK];
#pragma unroll
      for (int j = 0; j < 8; ++j) {
        fa0.u[j] = a0[lh * 8 + j]; fa0.u[8 + j] = a0[16 + lh * 8 + j];
        fa1.u[j] = a1[lh * 8 + j]; fa1.u[8 + j] = a1[16 + lh * 8 + j];
        fb0.u[j] = b0[lh * 8 + j]; fb0.u[8 + j] = b0[16 + lh * 8 + j];
        fb1.u[j] = b1[lh * 8 + j]; fb1.u[8 + j] = b1[16 + lh * 8 + j];
      }
    }

    acc00 = __builtin_amdgcn_wmma_f32_16x16x32_bf16(false, fa0.v, false, fb0.v, (short)0, acc00, false, false);
    acc01 = __builtin_amdgcn_wmma_f32_16x16x32_bf16(false, fa0.v, false, fb1.v, (short)0, acc01, false, false);
    acc10 = __builtin_amdgcn_wmma_f32_16x16x32_bf16(false, fa1.v, false, fb0.v, (short)0, acc10, false, false);
    acc11 = __builtin_amdgcn_wmma_f32_16x16x32_bf16(false, fa1.v, false, fb1.v, (short)0, acc11, false, false);
    __syncthreads();
  }

  // ---- epilogue: acc[r]: row = tileRow + r + lh*8, col = tileCol + (lane&15) ----
  int colA = col0 + wn * 32 + ml;        // tn = 0
  int colB = colA + 16;                  // tn = 1
  float bvA = (bias && colA < N) ? bias[colA] : 0.0f;
  float bvB = (bias && colB < N) ? bias[colB] : 0.0f;
#pragma unroll
  for (int r = 0; r < 8; ++r) {
    int rowA = row0 + wm * 32 + lh * 8 + r;   // tm = 0 (always < M)
    int rowB = rowA + 16;                     // tm = 1 (always < M)
    if (colA < N) {
      C[(size_t)rowA * ldc + colA] = apply_act(acc00[r] + bvA, act);
      C[(size_t)rowB * ldc + colA] = apply_act(acc10[r] + bvA, act);
    }
    if (colB < N) {
      C[(size_t)rowA * ldc + colB] = apply_act(acc01[r] + bvB, act);
      C[(size_t)rowB * ldc + colB] = apply_act(acc11[r] + bvB, act);
    }
  }
}

// ---------------- GAT helpers ----------------
// s = h@a_s, n = h@a_n (per-row reductions)
__global__ __launch_bounds__(256) void gat_sn_kernel(
    const float* __restrict__ h, const float* __restrict__ as,
    const float* __restrict__ an, float* __restrict__ s,
    float* __restrict__ nv, int F)
{
  __shared__ float r0[256], r1[256];
  int row = blockIdx.x;
  const float* hr = h + (size_t)row * F;
  float ps = 0.0f, pn = 0.0f;
  for (int f = threadIdx.x; f < F; f += 256) {
    float v = hr[f];
    ps += v * as[f];
    pn += v * an[f];
  }
  r0[threadIdx.x] = ps; r1[threadIdx.x] = pn;
  __syncthreads();
  for (int st = 128; st > 0; st >>= 1) {
    if (threadIdx.x < st) { r0[threadIdx.x] += r0[threadIdx.x + st];
                            r1[threadIdx.x] += r1[threadIdx.x + st]; }
    __syncthreads();
  }
  if (threadIdx.x == 0) { s[row] = r0[0]; nv[row] = r1[0]; }
}

// att[i,:] = softmax( mask(leaky((s_i + n_j) * M_ij)) )
__global__ __launch_bounds__(256) void gat_att_kernel(
    const float* __restrict__ s, const float* __restrict__ nv,
    const float* __restrict__ Mm, const float* __restrict__ adj,
    float* __restrict__ att)
{
  __shared__ float red[256];
  int row = blockIdx.x;
  const float* Mr = Mm  + (size_t)row * NN;
  const float* Ar = adj + (size_t)row * NN;
  float* Er = att + (size_t)row * NN;
  float si = s[row];

  float mx = -3.4e38f;
  for (int j = threadIdx.x; j < NN; j += 256) {
    float e = (si + nv[j]) * Mr[j];
    e = e > 0.0f ? e : 0.2f * e;            // leaky_relu alpha=0.2
    e = Ar[j] > 0.0f ? e : -9e15f;          // mask
    Er[j] = e;
    mx = fmaxf(mx, e);
  }
  red[threadIdx.x] = mx; __syncthreads();
  for (int st = 128; st > 0; st >>= 1) {
    if (threadIdx.x < st) red[threadIdx.x] = fmaxf(red[threadIdx.x], red[threadIdx.x + st]);
    __syncthreads();
  }
  mx = red[0]; __syncthreads();

  float sum = 0.0f;
  for (int j = threadIdx.x; j < NN; j += 256) {
    float w = expf(Er[j] - mx);
    Er[j] = w;
    sum += w;
  }
  red[threadIdx.x] = sum; __syncthreads();
  for (int st = 128; st > 0; st >>= 1) {
    if (threadIdx.x < st) red[threadIdx.x] += red[threadIdx.x + st];
    __syncthreads();
  }
  float inv = 1.0f / red[0];
  for (int j = threadIdx.x; j < NN; j += 256) Er[j] *= inv;
}

// ---------------- fusion gates ----------------
// m = l2norm(softmax(leaky01( concat(P,Q) @ W + b ))), W:[Fp+Fq,2]
__global__ __launch_bounds__(256) void gate2_kernel(
    const float* __restrict__ P, int Fp, const float* __restrict__ Q, int Fq,
    const float* __restrict__ W, const float* __restrict__ b,
    float* __restrict__ m)
{
  __shared__ float r0[256], r1[256];
  int row = blockIdx.x;
  const float* pr = P + (size_t)row * Fp;
  const float* qr = Q + (size_t)row * Fq;
  float d0 = 0.0f, d1 = 0.0f;
  for (int f = threadIdx.x; f < Fp; f += 256) {
    float v = pr[f]; d0 += v * W[2 * f]; d1 += v * W[2 * f + 1];
  }
  for (int f = threadIdx.x; f < Fq; f += 256) {
    float v = qr[f]; d0 += v * W[2 * (Fp + f)]; d1 += v * W[2 * (Fp + f) + 1];
  }
  r0[threadIdx.x] = d0; r1[threadIdx.x] = d1; __syncthreads();
  for (int st = 128; st > 0; st >>= 1) {
    if (threadIdx.x < st) { r0[threadIdx.x] += r0[threadIdx.x + st];
                            r1[threadIdx.x] += r1[threadIdx.x + st]; }
    __syncthreads();
  }
  if (threadIdx.x == 0) {
    float l0 = r0[0] + b[0], l1 = r1[0] + b[1];
    l0 = l0 > 0.0f ? l0 : 0.01f * l0;
    l1 = l1 > 0.0f ? l1 : 0.01f * l1;
    float mx = fmaxf(l0, l1);
    float e0 = expf(l0 - mx), e1 = expf(l1 - mx);
    float inv = 1.0f / (e0 + e1);
    float p0 = e0 * inv, p1 = e1 * inv;
    float nrm = fmaxf(sqrtf(p0 * p0 + p1 * p1), 1e-12f);
    m[(size_t)row * 2 + 0] = p0 / nrm;
    m[(size_t)row * 2 + 1] = p1 / nrm;
  }
}

// 5-way gate over concat(z1,z2,z3,z4,z), W:[3020,5]
__global__ __launch_bounds__(256) void gate5_kernel(
    const float* __restrict__ z1, const float* __restrict__ z2,
    const float* __restrict__ z3, const float* __restrict__ z4,
    const float* __restrict__ z, const float* __restrict__ W,
    const float* __restrict__ b, float* __restrict__ u)
{
  __shared__ float red[5][256];
  int row = blockIdx.x;
  float d[5] = {0, 0, 0, 0, 0};
  for (int idx = threadIdx.x; idx < 3020; idx += 256) {
    float v;
    if (idx < 500)       v = z1[(size_t)row * 500 + idx];
    else if (idx < 1000) v = z2[(size_t)row * 500 + (idx - 500)];
    else if (idx < 3000) v = z3[(size_t)row * 2000 + (idx - 1000)];
    else if (idx < 3010) v = z4[(size_t)row * 10 + (idx - 3000)];
    else                 v = z[(size_t)row * 10 + (idx - 3010)];
#pragma unroll
    for (int c = 0; c < 5; ++c) d[c] += v * W[(size_t)idx * 5 + c];
  }
#pragma unroll
  for (int c = 0; c < 5; ++c) red[c][threadIdx.x] = d[c];
  __syncthreads();
  for (int st = 128; st > 0; st >>= 1) {
    if (threadIdx.x < st)
      for (int c = 0; c < 5; ++c) red[c][threadIdx.x] += red[c][threadIdx.x + st];
    __syncthreads();
  }
  if (threadIdx.x == 0) {
    float l[5], mx = -3.4e38f;
    for (int c = 0; c < 5; ++c) {
      l[c] = red[c][0] + b[c];
      l[c] = l[c] > 0.0f ? l[c] : 0.01f * l[c];
      mx = fmaxf(mx, l[c]);
    }
    float s5 = 0.0f;
    for (int c = 0; c < 5; ++c) { l[c] = expf(l[c] - mx); s5 += l[c]; }
    float inv = 1.0f / s5, nrm = 0.0f;
    for (int c = 0; c < 5; ++c) { l[c] *= inv; nrm += l[c] * l[c]; }
    nrm = fmaxf(sqrtf(nrm), 1e-12f);
    for (int c = 0; c < 5; ++c) u[(size_t)row * 5 + c] = l[c] / nrm;
  }
}

// out = m[:,0]*A + m[:,1]*B (row-broadcast)
__global__ void mix2_kernel(const float* __restrict__ m, const float* __restrict__ A,
                            const float* __restrict__ B, float* __restrict__ out, int F)
{
  size_t i = (size_t)blockIdx.x * 256 + threadIdx.x;
  size_t total = (size_t)NN * F;
  if (i >= total) return;
  size_t row = i / F;
  out[i] = m[row * 2] * A[i] + m[row * 2 + 1] * B[i];
}

// fused = concat(u0*z1, u1*z2, u2*z3, u3*z4, u4*z) : [4096,3020]
__global__ void fuse5_kernel(const float* __restrict__ u,
                             const float* __restrict__ z1, const float* __restrict__ z2,
                             const float* __restrict__ z3, const float* __restrict__ z4,
                             const float* __restrict__ z, float* __restrict__ fused)
{
  size_t i = (size_t)blockIdx.x * 256 + threadIdx.x;
  size_t total = (size_t)NN * 3020;
  if (i >= total) return;
  size_t row = i / 3020;
  int col = (int)(i % 3020);
  const float* ur = u + row * 5;
  float v;
  if (col < 500)       v = ur[0] * z1[row * 500 + col];
  else if (col < 1000) v = ur[1] * z2[row * 500 + (col - 500)];
  else if (col < 3000) v = ur[2] * z3[row * 2000 + (col - 1000)];
  else if (col < 3010) v = ur[3] * z4[row * 10 + (col - 3000)];
  else                 v = ur[4] * z[row * 10 + (col - 3010)];
  fused[i] = v;
}

__global__ void softmax_rows_kernel(const float* __restrict__ in, float* __restrict__ out,
                                    int rows, int w)
{
  int r = blockIdx.x * 256 + threadIdx.x;
  if (r >= rows) return;
  const float* x = in + (size_t)r * w;
  float* o = out + (size_t)r * w;
  float mx = x[0];
  for (int j = 1; j < w; ++j) mx = fmaxf(mx, x[j]);
  float s = 0.0f;
  for (int j = 0; j < w; ++j) { float e = expf(x[j] - mx); o[j] = e; s += e; }
  float inv = 1.0f / s;
  for (int j = 0; j < w; ++j) o[j] *= inv;
}

// Student-t q (V=1 -> exponent 1): q_c = (1/(1+d2)) normalized
__global__ void q_kernel(const float* __restrict__ Z, const float* __restrict__ cl,
                         float* __restrict__ q, int rows)
{
  int r = blockIdx.x * 256 + threadIdx.x;
  if (r >= rows) return;
  const float* z = Z + (size_t)r * NZD;
  float qq[NCD], s = 0.0f;
  for (int c = 0; c < NCD; ++c) {
    float d2 = 0.0f;
    for (int k = 0; k < NZD; ++k) { float t = z[k] - cl[c * NZD + k]; d2 += t * t; }
    float v = 1.0f / (1.0f + d2);
    qq[c] = v; s += v;
  }
  float inv = 1.0f / s;
  for (int c = 0; c < NCD; ++c) q[(size_t)r * NCD + c] = qq[c] * inv;
}

// ---------------- host orchestration ----------------
static void launch_gemm(hipStream_t st, const float* A, const float* B, const float* bias,
                        float* C, int M, int N, int K, int transB, int act)
{
  dim3 grid((N + BN - 1) / BN, (M + BM - 1) / BM);
  gemm_bf16_wmma<<<grid, 256, 0, st>>>(A, B, bias, C, M, N, K, transB ? K : N, transB, act);
}

static void run_gat(hipStream_t st, const float* X, int Fin,
                    const float* W, const float* as, const float* an, int Fout,
                    const float* adj, const float* Mm,
                    float* hg, float* att, float* s, float* nv, float* Zout)
{
  launch_gemm(st, X, W, nullptr, hg, NN, Fout, Fin, 0, ACT_NONE);
  gat_sn_kernel<<<NN, 256, 0, st>>>(hg, as, an, s, nv, Fout);
  gat_att_kernel<<<NN, 256, 0, st>>>(s, nv, Mm, adj, att);
  launch_gemm(st, att, hg, nullptr, Zout, NN, Fout, NN, 0, ACT_ELU);
}

extern "C" void kernel_launch(void* const* d_in, const int* in_sizes, int n_in,
                              void* d_out, int out_size, void* d_ws, size_t ws_size,
                              hipStream_t stream)
{
  (void)in_sizes; (void)n_in; (void)out_size; (void)ws_size;
  const float* x    = (const float*)d_in[0];
  const float* adj  = (const float*)d_in[1];
  const float* Mm   = (const float*)d_in[2];
  const float* e1W  = (const float*)d_in[3];  const float* e1b  = (const float*)d_in[4];
  const float* e2W  = (const float*)d_in[5];  const float* e2b  = (const float*)d_in[6];
  const float* e3W  = (const float*)d_in[7];  const float* e3b  = (const float*)d_in[8];
  const float* zlW  = (const float*)d_in[9];  const float* zlb  = (const float*)d_in[10];
  const float* d1W  = (const float*)d_in[11]; const float* d1b  = (const float*)d_in[12];
  const float* d2W  = (const float*)d_in[13]; const float* d2b  = (const float*)d_in[14];
  const float* d3W  = (const float*)d_in[15]; const float* d3b  = (const float*)d_in[16];
  const float* xbW  = (const float*)d_in[17]; const float* xbb  = (const float*)d_in[18];
  const float* g0W  = (const float*)d_in[19]; const float* g0s  = (const float*)d_in[20]; const float* g0n = (const float*)d_in[21];
  const float* g1W  = (const float*)d_in[22]; const float* g1s  = (const float*)d_in[23]; const float* g1n = (const float*)d_in[24];
  const float* g2W  = (const float*)d_in[25]; const float* g2s  = (const float*)d_in[26]; const float* g2n = (const float*)d_in[27];
  const float* g3W  = (const float*)d_in[28]; const float* g3s  = (const float*)d_in[29]; const float* g3n = (const float*)d_in[30];
  const float* dg1W = (const float*)d_in[31]; const float* dg1s = (const float*)d_in[32]; const float* dg1n = (const float*)d_in[33];
  const float* dg2W = (const float*)d_in[34]; const float* dg2s = (const float*)d_in[35]; const float* dg2n = (const float*)d_in[36];
  const float* dg3W = (const float*)d_in[37]; const float* dg3s = (const float*)d_in[38]; const float* dg3n = (const float*)d_in[39];
  const float* gz   = (const float*)d_in[40];
  const float* F1W  = (const float*)d_in[41]; const float* F1b  = (const float*)d_in[42];
  const float* F2W  = (const float*)d_in[43]; const float* F2b  = (const float*)d_in[44];
  const float* F3W  = (const float*)d_in[45]; const float* F3b  = (const float*)d_in[46];
  const float* FLW  = (const float*)d_in[47]; const float* FLb  = (const float*)d_in[48];
  const float* clus = (const float*)d_in[49];

  float* out = (float*)d_out;
  float* ws  = (float*)d_ws;

  // output offsets (floats)
  const size_t O_XBAR = 0;
  const size_t O_QAE  = O_XBAR + (size_t)NN * N_IN;
  const size_t O_QGAT = O_QAE  + (size_t)NN * NCD;
  const size_t O_PRED = O_QGAT + (size_t)NN * NCD;
  const size_t O_Z    = O_PRED + (size_t)NN * NCD;
  const size_t O_NET  = O_Z    + (size_t)NN * NZD;
  const size_t O_APRD = O_NET  + (size_t)NN * NCD;
  const size_t O_ZGAT = O_APRD + (size_t)NN * NN;
  const size_t O_AGAT = O_ZGAT + (size_t)NN * N_IN;

  // workspace layout (floats)
  size_t off = 0;
  float* ATT = ws + off; off += (size_t)NN * NN;      // attention / fused scratch
  float* HG  = ws + off; off += (size_t)NN * N_IN;    // GAT h buffer (<=3000 wide)
  float* H1  = ws + off; off += (size_t)NN * E1D;
  float* H2  = ws + off; off += (size_t)NN * E2D;
  float* H3  = ws + off; off += (size_t)NN * E3D;
  float* T1  = ws + off; off += (size_t)NN * E3D;     // 2000-wide temp
  float* T2  = ws + off; off += (size_t)NN * E2D;     // 500-wide temp
  float* Z1  = ws + off; off += (size_t)NN * E1D;
  float* Z2  = ws + off; off += (size_t)NN * E2D;
  float* Z3  = ws + off; off += (size_t)NN * E3D;
  float* Z4  = ws + off; off += (size_t)NN * NZD;
  float* Sv  = ws + off; off += NN;
  float* Nv  = ws + off; off += NN;
  float* M2  = ws + off; off += (size_t)NN * 2;
  float* Uv  = ws + off; off += (size_t)NN * 5;
  float* Tt  = ws + off; off += (size_t)NN * NCD;

  float* Zout = out + O_Z;

  // ---- autoencoder ----
  launch_gemm(stream, x,  e1W, e1b, H1, NN, E1D, N_IN, 0, ACT_RELU);
  launch_gemm(stream, H1, e2W, e2b, H2, NN, E2D, E1D, 0, ACT_RELU);
  launch_gemm(stream, H2, e3W, e3b, H3, NN, E3D, E2D, 0, ACT_RELU);
  launch_gemm(stream, H3, zlW, zlb, Zout, NN, NZD, E3D, 0, ACT_NONE);
  launch_gemm(stream, Zout, d1W, d1b, T1, NN, E3D, NZD, 0, ACT_RELU);
  launch_gemm(stream, T1, d2W, d2b, T2, NN, E2D, E3D, 0, ACT_RELU);
  launch_gemm(stream, T2, d3W, d3b, T1, NN, E1D, E2D, 0, ACT_RELU);   // d3 reuses T1
  launch_gemm(stream, T1, xbW, xbb, out + O_XBAR, NN, N_IN, E1D, 0, ACT_NONE);

  // ---- GAT encoder chain ----
  run_gat(stream, x, N_IN, g0W, g0s, g0n, E1D, adj, Mm, HG, ATT, Sv, Nv, Z1);
  gate2_kernel<<<NN, 256, 0, stream>>>(H1, E1D, Z1, E1D, F1W, F1b, M2);
  {
    size_t tot = (size_t)NN * E1D;
    mix2_kernel<<<(unsigned)((tot + 255) / 256), 256, 0, stream>>>(M2, Z1, H1, T2, E1D); // net1
  }
  run_gat(stream, T2, E1D, g1W, g1s, g1n, E2D, adj, Mm, HG, ATT, Sv, Nv, Z2);
  gate2_kernel<<<NN, 256, 0, stream>>>(H2, E2D, Z2, E2D, F2W, F2b, M2);
  {
    size_t tot = (size_t)NN * E2D;
    mix2_kernel<<<(unsigned)((tot + 255) / 256), 256, 0, stream>>>(M2, H2, Z2, T2, E2D); // net2
  }
  run_gat(stream, T2, E2D, g2W, g2s, g2n, E3D, adj, Mm, HG, ATT, Sv, Nv, Z3);
  gate2_kernel<<<NN, 256, 0, stream>>>(H3, E3D, Z3, E3D, F3W, F3b, M2);
  {
    size_t tot = (size_t)NN * E3D;
    mix2_kernel<<<(unsigned)((tot + 255) / 256), 256, 0, stream>>>(M2, H3, Z3, T1, E3D); // net3
  }
  run_gat(stream, T1, E3D, g3W, g3s, g3n, NZD, adj, Mm, HG, ATT, Sv, Nv, Z4);

  // A_pred = sigmoid(z4 @ z4.T)
  launch_gemm(stream, Z4, Z4, nullptr, out + O_APRD, NN, NN, NZD, 1, ACT_SIG);

  // ---- 5-way fusion head (uses ATT as the fused buffer before decoder GATs) ----
  gate5_kernel<<<NN, 256, 0, stream>>>(Z1, Z2, Z3, Z4, Zout, FLW, FLb, Uv);
  {
    size_t tot = (size_t)NN * 3020;
    fuse5_kernel<<<(unsigned)((tot + 255) / 256), 256, 0, stream>>>(Uv, Z1, Z2, Z3, Z4, Zout, ATT);
  }
  launch_gemm(stream, ATT, gz, nullptr, Tt, NN, NCD, 3020, 0, ACT_NONE);
  launch_gemm(stream, adj, Tt, nullptr, out + O_NET, NN, NCD, NN, 0, ACT_NONE);
  softmax_rows_kernel<<<(NN + 255) / 256, 256, 0, stream>>>(out + O_NET, out + O_PRED, NN, NCD);

  // ---- q distributions ----
  q_kernel<<<(NN + 255) / 256, 256, 0, stream>>>(Zout, clus, out + O_QAE, NN);
  q_kernel<<<(NN + 255) / 256, 256, 0, stream>>>(Z4,   clus, out + O_QGAT, NN);

  // ---- GAT decoder chain ----
  run_gat(stream, Z4, NZD, dg1W, dg1s, dg1n, E3D, adj, Mm, HG, ATT, Sv, Nv, T1);   // zg1
  run_gat(stream, T1, E3D, dg2W, dg2s, dg2n, E2D, adj, Mm, HG, ATT, Sv, Nv, T2);   // zg2
  run_gat(stream, T2, E2D, dg3W, dg3s, dg3n, N_IN, adj, Mm, HG, ATT, Sv, Nv, out + O_ZGAT);
  launch_gemm(stream, out + O_ZGAT, out + O_ZGAT, nullptr, out + O_AGAT, NN, NN, N_IN, 1, ACT_SIG);
}
